// MCAttEGNN_69088843923955
// MI455X (gfx1250) — matrix operand order repaired and checked by reading, not compiled, not run on hardware
//
#include <hip/hip_runtime.h>
#include <hip/hip_bf16.h>

// ---------------------------------------------------------------------------
// MCAttEGNN on MI455X (gfx1250): f16 WMMA GEMMs, TDM gather-mode feed of the
// target_feat LDS tile, atomic segment softmax.
// ---------------------------------------------------------------------------

typedef __attribute__((ext_vector_type(16))) _Float16 v16h;
typedef __attribute__((ext_vector_type(8)))  float    v8f;
typedef __attribute__((ext_vector_type(4)))  unsigned uivec4;
typedef __attribute__((ext_vector_type(8)))  int      ivec8;
typedef __attribute__((ext_vector_type(4)))  int      ivec4;

union ABu { uint4 u4[2]; v16h v; };

#define Dm   128   // node feature dim
#define KVIN 160   // 16 radial + 128 h[col] + 16 edge_attr
#define HSD  512   // HS * D
#define TILE 16

// Per-block (1 wave) LDS working set for one 16-edge tile
struct alignas(16) WaveSmem {
  _Float16 feat[16 * KVIN];  // target_feat tile (f16)          5120 B
  _Float16 qls [16 * Dm];    // q tile (f16, +bias)             4096 B
  _Float16 vls [16 * Dm];    // v tile (f16, +bias)             4096 B
  _Float16 sls [16 * 256];   // silu half-tile (f16)            8192 B
  float    alph[16];         // per-edge alpha accumulators       64 B
};                           // total 21568 B

__device__ inline void atomicMaxFloat(float* addr, float v) {
  // works with init bits 0xFF800000 (-inf)
  if (v >= 0.f) atomicMax((int*)addr, __float_as_int(v));
  else          atomicMin((unsigned int*)addr, (unsigned int)__float_as_int(v));
}

// ----------------------------- prep kernels --------------------------------

__global__ void k_fill_f32(float* x, float v, long n) {
  long i = (long)blockIdx.x * blockDim.x + threadIdx.x;
  if (i < n) x[i] = v;
}

__global__ void k_f32_to_f16(const float* __restrict__ x, _Float16* __restrict__ y, long n) {
  long i = (long)blockIdx.x * blockDim.x + threadIdx.x;
  if (i < n) y[i] = (_Float16)x[i];
}

// W is [K][Nc] row-major (f32); WT is [Nc][K] row-major (f16)  (B^T layout)
__global__ void k_transpose_f16(const float* __restrict__ W, _Float16* __restrict__ WT,
                                int K, int Nc) {
  long i = (long)blockIdx.x * blockDim.x + threadIdx.x;
  if (i >= (long)K * Nc) return;
  int n = (int)(i / K), k = (int)(i % K);
  WT[(long)n * K + k] = (_Float16)W[(long)k * Nc + n];
}

// Wc2 is [512][4]; pad to B^T [16][512] with zero rows 4..15
__global__ void k_wc2_pad(const float* __restrict__ W, _Float16* __restrict__ WT) {
  int i = blockIdx.x * blockDim.x + threadIdx.x;
  if (i >= 16 * HSD) return;
  int n = i / HSD, k = i % HSD;
  WT[i] = (n < 4) ? (_Float16)W[k * 4 + n] : (_Float16)0.f;
}

// ------------------------- edge tile kernel (WMMA + TDM) -------------------

__global__ void __launch_bounds__(32)
k_edge_tiles(const float* __restrict__ coord,
             const int* __restrict__ erow, const int* __restrict__ ecol,
             const float* __restrict__ edge_attr,
             const float* __restrict__ bq, const float* __restrict__ bkv,
             const _Float16* __restrict__ h16,
             const _Float16* __restrict__ WqT,  const _Float16* __restrict__ WkvT,
             const _Float16* __restrict__ Wc1T, const _Float16* __restrict__ Wc2T,
             float* __restrict__ alpha_g, _Float16* __restrict__ vbuf,
             float* __restrict__ cv_g, float* __restrict__ cdiff_g,
             float* __restrict__ mbuf, long Nn, long E)
{
  extern __shared__ char raw[];
  WaveSmem* sm = (WaveSmem*)raw;
  const int lane = threadIdx.x & 31;
  const int lo   = lane & 15;     // A row / B,C,D column within fragment
  const int hf   = lane >> 4;     // lane-half selector

  const long eBase = (long)blockIdx.x * TILE;   // uniform per wave
  const long eMine = eBase + lo;
  const bool vMine = eMine < E;
  const long eCl   = vMine ? eMine : (E - 1);

  const int rnode = erow[eCl];
  const int cnode = ecol[eCl];

  // ---- TDM gather: h16[col[e]] rows -> feat[:, 16..143] (async, DMA) ----
  // D# per CDNA5 ISA: gather mode, 16x 16-bit row
  // indices, data_size=2B, tile rows of 128 elems (256B = 64 DWORDs), LDS
  // padding of 16 DWORDs (64B) after every row to match the feat stride.
  {
    unsigned idxp[8];
#pragma unroll
    for (int i = 0; i < 8; i++) {                // uniform scalar loads
      long e0 = eBase + 2 * i;
      long e1 = e0 + 1;
      unsigned i0 = (unsigned)ecol[e0 < E ? e0 : (E - 1)];
      unsigned i1 = (unsigned)ecol[e1 < E ? e1 : (E - 1)];
      idxp[i] = (i0 & 0xFFFFu) | ((i1 & 0xFFFFu) << 16);
    }
    unsigned long long gaddr = (unsigned long long)(uintptr_t)h16;
    // low 32 bits of a flat shared pointer == LDS byte offset (LDS aperture)
    unsigned ldsDst = (unsigned)(uintptr_t)(void*)(sm->feat + 16);

    uivec4 g0;
    g0[0] = 0x80000001u;                               // count=1 | gather_mode, 16b idx
    g0[1] = ldsDst;                                    // lds_addr
    g0[2] = (unsigned)(gaddr & 0xFFFFFFFFu);           // global_addr[31:0]
    g0[3] = (unsigned)((gaddr >> 32) & 0x01FFFFFFu)    // global_addr[56:32]
          | (2u << 30);                                // type = 2 ("image")

    ivec8 g1;
    g1[0] = (int)((1u << 16)        // data_size = 2 bytes
                | (1u << 20)        // pad_enable
                | (5u << 22)        // pad_interval code 5 = 64 DWORDs
                | (15u << 25));     // pad_amount  code 15 = 16 DWORDs
    g1[1] = (int)((128u & 0xFFFFu) << 16);                       // tensor_dim0 lo16
    g1[2] = (int)(((unsigned)Nn & 0xFFFFu) << 16);               // dim0 hi16 | tensor_dim1 lo16
    g1[3] = (int)((((unsigned)Nn >> 16) & 0xFFFFu) | (128u << 16)); // dim1 hi16 | tile_dim0=128
    g1[4] = (int)16u;                                            // tile_dim1 = 16 indices
    g1[5] = (int)128u;                                           // tensor_dim0_stride = 128
    g1[6] = 0;
    g1[7] = 0;

    ivec4 g2 = { (int)idxp[0], (int)idxp[1], (int)idxp[2], (int)idxp[3] };
    ivec4 g3 = { (int)idxp[4], (int)idxp[5], (int)idxp[6], (int)idxp[7] };
    ivec8 gz = { 0, 0, 0, 0, 0, 0, 0, 0 };             // unused trailing group

    __builtin_amdgcn_tensor_load_to_lds(g0, g1, g2, g3, gz, 0 /*cpol*/);
  }

  // keep weight streams warm in WGP$/L2
  __builtin_prefetch(WqT, 0, 3);
  __builtin_prefetch(WkvT, 0, 3);
  __builtin_prefetch(Wc1T, 0, 3);

  // ---- stage 1: scalar per-edge prep (lanes 0..15 own one edge each) ----
  if (lane < 16) {
    sm->alph[lo] = 0.f;
    float dif[12];
    const float* cr = coord + (long)rnode * 12;
    const float* cc = coord + (long)cnode * 12;
#pragma unroll
    for (int i = 0; i < 12; i++) dif[i] = cr[i] - cc[i];
#pragma unroll
    for (int i = 0; i < 4; i++)
#pragma unroll
      for (int j = 0; j < 4; j++) {
        float rd = dif[i*3]*dif[j*3] + dif[i*3+1]*dif[j*3+1] + dif[i*3+2]*dif[j*3+2];
        sm->feat[lo * KVIN + i * 4 + j] = (_Float16)rd;
      }
    const float* ea = edge_attr + eCl * 16;
#pragma unroll
    for (int k = 0; k < 16; k++) sm->feat[lo * KVIN + 144 + k] = (_Float16)ea[k];
    if (vMine) {
#pragma unroll
      for (int i = 0; i < 12; i++) cdiff_g[eMine * 12 + i] = dif[i];
    }
  }
  __syncthreads();   // radial/edge_attr in feat; TDM still in flight

  // ---- stage 2: q = h[row] @ Wq + bq  (overlaps with TDM gather) ----
  const _Float16* hrow = h16 + (long)rnode * Dm;
  for (int jq = 0; jq < 8; jq++) {
    v8f acc = {};
#pragma unroll
    for (int ks = 0; ks < 4; ks++) {
      ABu a, b;
      a.u4[0] = *(const uint4*)(hrow + ks * 32 + 8 * hf);
      a.u4[1] = *(const uint4*)(hrow + ks * 32 + 16 + 8 * hf);
      const _Float16* bp = WqT + (long)(jq * 16 + lo) * Dm + ks * 32 + 8 * hf;
      b.u4[0] = *(const uint4*)(bp);
      b.u4[1] = *(const uint4*)(bp + 16);
      acc = __builtin_amdgcn_wmma_f32_16x16x32_f16(false, a.v, false, b.v,
                                                   (short)0, acc, false, false);
    }
    float bqv = bq[jq * 16 + lo];
#pragma unroll
    for (int r = 0; r < 8; r++)
      sm->qls[(r + 8 * hf) * Dm + jq * 16 + lo] = (_Float16)(acc[r] + bqv);
  }
  __builtin_amdgcn_s_wait_tensorcnt(0);   // feat h-part landed in LDS
  __syncthreads();

  // ---- stage 3: kv = feat @ Wkv + bkv; de-interleave k/v; alpha = sum q*k ----
  float pal[8];
#pragma unroll
  for (int r = 0; r < 8; r++) pal[r] = 0.f;
  for (int jk = 0; jk < 16; jk++) {
    v8f acc = {};
#pragma unroll
    for (int ks = 0; ks < 5; ks++) {
      ABu a, b;
      const _Float16* ap = sm->feat + lo * KVIN + ks * 32 + 8 * hf;
      a.u4[0] = *(const uint4*)(ap);
      a.u4[1] = *(const uint4*)(ap + 16);
      const _Float16* bp = WkvT + (long)(jk * 16 + lo) * KVIN + ks * 32 + 8 * hf;
      b.u4[0] = *(const uint4*)(bp);
      b.u4[1] = *(const uint4*)(bp + 16);
      acc = __builtin_amdgcn_wmma_f32_16x16x32_f16(false, a.v, false, b.v,
                                                   (short)0, acc, false, false);
    }
    const int   cg = jk * 16 + lo;     // global kv column
    const float bv = bkv[cg];
    const int   t  = cg >> 1;
    if ((cg & 1) == 0) {               // k column: alpha partials vs q (LDS)
#pragma unroll
      for (int r = 0; r < 8; r++)
        pal[r] += (acc[r] + bv) * (float)sm->qls[(r + 8 * hf) * Dm + t];
    } else {                           // v column: stash f16 tile
#pragma unroll
      for (int r = 0; r < 8; r++)
        sm->vls[(r + 8 * hf) * Dm + t] = (_Float16)(acc[r] + bv);
    }
  }
  if ((lo & 1) == 0) {
#pragma unroll
    for (int r = 0; r < 8; r++) atomicAdd(&sm->alph[r + 8 * hf], pal[r]);  // ds_add_f32
  }
  __syncthreads();

  // ---- stage 4: alpha out + segment max; spill v to global (f16) ----
  if (lane < 16 && vMine) {
    float av = sm->alph[lo];
    alpha_g[eMine] = av;
    atomicMaxFloat(mbuf + rnode, av);
  }
  if (vMine) {
    const uint4* src = (const uint4*)(sm->vls + lo * Dm + hf * 64);
    uint4*       dst = (uint4*)(vbuf + eMine * Dm + hf * 64);
#pragma unroll
    for (int t = 0; t < 8; t++) dst[t] = src[t];
  }

  // ---- stage 5: cv = silu(v @ Wc1) @ Wc2  (two 256-col halves) ----
  v8f cvacc = {};
  for (int half = 0; half < 2; half++) {
    for (int jn = 0; jn < 16; jn++) {
      v8f acc = {};
#pragma unroll
      for (int ks = 0; ks < 4; ks++) {
        ABu a, b;
        const _Float16* ap = sm->vls + lo * Dm + ks * 32 + 8 * hf;
        a.u4[0] = *(const uint4*)(ap);
        a.u4[1] = *(const uint4*)(ap + 16);
        const _Float16* bp = Wc1T + (long)(half * 256 + jn * 16 + lo) * Dm + ks * 32 + 8 * hf;
        b.u4[0] = *(const uint4*)(bp);
        b.u4[1] = *(const uint4*)(bp + 16);
        acc = __builtin_amdgcn_wmma_f32_16x16x32_f16(false, a.v, false, b.v,
                                                     (short)0, acc, false, false);
      }
#pragma unroll
      for (int r = 0; r < 8; r++) {
        float x = acc[r];
        float s = x / (1.f + __expf(-x));                 // silu
        sm->sls[(r + 8 * hf) * 256 + jn * 16 + lo] = (_Float16)s;
      }
    }
    __syncthreads();
#pragma unroll
    for (int ks = 0; ks < 8; ks++) {
      ABu a, b;
      const _Float16* ap = sm->sls + lo * 256 + ks * 32 + 8 * hf;
      a.u4[0] = *(const uint4*)(ap);
      a.u4[1] = *(const uint4*)(ap + 16);
      const _Float16* bp = Wc2T + (long)lo * HSD + half * 256 + ks * 32 + 8 * hf;
      b.u4[0] = *(const uint4*)(bp);
      b.u4[1] = *(const uint4*)(bp + 16);
      cvacc = __builtin_amdgcn_wmma_f32_16x16x32_f16(false, a.v, false, b.v,
                                                     (short)0, cvacc, false, false);
    }
    __syncthreads();
  }
  if (lo < 4) {                        // only cols 0..3 of padded Wc2 are live
#pragma unroll
    for (int r = 0; r < 8; r++) {
      long e = eBase + r + 8 * hf;
      if (e < E) cv_g[e * 4 + lo] = cvacc[r];
    }
  }
}

// -------------------------- softmax / scatter ------------------------------

__global__ void k_exp_sum(const float* __restrict__ alpha_g, const int* __restrict__ erow,
                          const float* __restrict__ mbuf, float* __restrict__ expe,
                          float* __restrict__ sbuf, long E) {
  long e = (long)blockIdx.x * blockDim.x + threadIdx.x;
  if (e >= E) return;
  int r = erow[e];
  float ex = __expf(alpha_g[e] - mbuf[r]);
  expe[e] = ex;
  atomicAdd(sbuf + r, ex);
}

__global__ void k_scatter(const int* __restrict__ erow, const float* __restrict__ expe,
                          const float* __restrict__ sbuf, const _Float16* __restrict__ vbuf,
                          const float* __restrict__ cv_g, const float* __restrict__ cdiff_g,
                          float* __restrict__ agg, float* __restrict__ cagg,
                          float* __restrict__ att_out, long E) {
  const int lane = threadIdx.x & 31;
  long e = (long)blockIdx.x * 8 + (threadIdx.x >> 5);  // 1 wave per edge
  if (e >= E) return;
  int r = erow[e];
  float att = expe[e] / sbuf[r];
  if (lane == 0) att_out[e] = att;
#pragma unroll
  for (int t = 0; t < 4; t++) {
    int col = lane + t * 32;
    atomicAdd(agg + (long)r * Dm + col, att * (float)vbuf[e * Dm + col]);
  }
  if (lane < 12) {
    float tr = att * cv_g[e * 4 + lane / 3] * cdiff_g[e * 12 + lane];
    atomicAdd(cagg + (long)r * 12 + lane, tr);
  }
}

__global__ void k_finalize(const float* __restrict__ h, const float* __restrict__ coord,
                           const float* __restrict__ agg, const float* __restrict__ cagg,
                           float* __restrict__ out_h, float* __restrict__ out_c,
                           long nh, long nc) {
  long i = (long)blockIdx.x * blockDim.x + threadIdx.x;
  if (i < nh) {
    out_h[i] = h[i] + agg[i];
  } else if (i < nh + nc) {
    long j = i - nh;
    float v = cagg[j];
    v = fminf(fmaxf(v, -10.f), 10.f);
    out_c[j] = coord[j] + v;
  }
}

// ------------------------------- launcher ----------------------------------

static inline size_t alignUp(size_t x, size_t a) { return (x + a - 1) & ~(a - 1); }

extern "C" void kernel_launch(void* const* d_in, const int* in_sizes, int n_in,
                              void* d_out, int out_size, void* d_ws, size_t ws_size,
                              hipStream_t stream) {
  const float* h    = (const float*)d_in[0];
  const float* crd  = (const float*)d_in[1];
  const int*   eidx = (const int*)d_in[2];
  const float* ea   = (const float*)d_in[3];
  const float* Wq   = (const float*)d_in[4];
  const float* bq   = (const float*)d_in[5];
  const float* Wkv  = (const float*)d_in[6];
  const float* bkv  = (const float*)d_in[7];
  const float* Wc1  = (const float*)d_in[8];
  const float* Wc2  = (const float*)d_in[9];

  const long N = in_sizes[0] / Dm;
  const long E = in_sizes[2] / 2;
  const int* erow = eidx;
  const int* ecol = eidx + E;

  // workspace carve-up
  char* ws = (char*)d_ws; size_t off = 0;
  auto take = [&](size_t bytes) { char* p = ws + off; off = alignUp(off + bytes, 256); return p; };
  _Float16* h16    = (_Float16*)take((size_t)N * Dm * 2);
  _Float16* WqT    = (_Float16*)take((size_t)Dm * Dm * 2);
  _Float16* WkvT   = (_Float16*)take((size_t)256 * KVIN * 2);
  _Float16* Wc1T   = (_Float16*)take((size_t)HSD * Dm * 2);
  _Float16* Wc2T   = (_Float16*)take((size_t)16 * HSD * 2);
  float*    alpha  = (float*)take((size_t)E * 4);
  float*    expe   = (float*)take((size_t)E * 4);
  _Float16* vbuf   = (_Float16*)take((size_t)E * Dm * 2);
  float*    cv_g   = (float*)take((size_t)E * 4 * 4);
  float*    cdiff  = (float*)take((size_t)E * 12 * 4);
  float*    mbuf   = (float*)take((size_t)N * 4);
  float*    sbuf   = (float*)take((size_t)N * 4);
  float*    agg    = (float*)take((size_t)N * Dm * 4);
  float*    cagg   = (float*)take((size_t)N * 12 * 4);

  float* out_h   = (float*)d_out;
  float* out_c   = out_h + N * Dm;
  float* out_att = out_c + N * 12;

  // init accumulators (memset nodes are graph-capturable)
  (void)hipMemsetAsync(sbuf, 0, (size_t)N * 4, stream);
  (void)hipMemsetAsync(agg,  0, (size_t)N * Dm * 4, stream);
  (void)hipMemsetAsync(cagg, 0, (size_t)N * 12 * 4, stream);
  {
    long n = N;
    k_fill_f32<<<(n + 255) / 256, 256, 0, stream>>>(mbuf, -INFINITY, n);
  }
  {
    long n = N * Dm;
    k_f32_to_f16<<<(n + 255) / 256, 256, 0, stream>>>(h, h16, n);
  }
  k_transpose_f16<<<(Dm * Dm + 255) / 256, 256, 0, stream>>>(Wq, WqT, Dm, Dm);
  k_transpose_f16<<<(KVIN * 256 + 255) / 256, 256, 0, stream>>>(Wkv, WkvT, KVIN, 256);
  k_transpose_f16<<<(Dm * HSD + 255) / 256, 256, 0, stream>>>(Wc1, Wc1T, Dm, HSD);
  k_wc2_pad<<<(16 * HSD + 255) / 256, 256, 0, stream>>>(Wc2, Wc2T);

  // main edge kernel: 1 wave (block of 32) per 16-edge tile
  const long numTiles = (E + TILE - 1) / TILE;
  const size_t dynLds = sizeof(WaveSmem);
  k_edge_tiles<<<dim3((unsigned)numTiles), dim3(32), dynLds, stream>>>(
      crd, erow, ecol, ea, bq, bkv, h16, WqT, WkvT, Wc1T, Wc2T,
      alpha, vbuf, cv_g, cdiff, mbuf, N, E);

  k_exp_sum<<<(unsigned)((E + 255) / 256), 256, 0, stream>>>(alpha, erow, mbuf, expe, sbuf, E);

  k_scatter<<<(unsigned)((E + 7) / 8), 256, 0, stream>>>(erow, expe, sbuf, vbuf, cv_g,
                                                         cdiff, agg, cagg, out_att, E);

  const long nh = N * Dm, nc = N * 12;
  k_finalize<<<(unsigned)((nh + nc + 255) / 256), 256, 0, stream>>>(h, crd, agg, cagg,
                                                                    out_h, out_c, nh, nc);
}
